// CrossLayerTranscoder_38354057953757
// MI455X (gfx1250) — matrix operand order repaired and checked by reading, not compiled, not run on hardware
//
#include <hip/hip_runtime.h>

// ---------------------------------------------------------------------------
// Cross-layer transcoder for MI455X (gfx1250): bf16 WMMA, f32 accumulate.
//   feats[l] = relu(resid[l] @ enc_w[l]^T)          encode kernel
//   recon[t] = sum_{s<=t} feats[s] @ dec_w[s,t]^T   decode kernel
// Double-buffered LDS + register prefetch pipeline, one barrier per K-step.
// f32->bf16 staging via v_add + v_perm_b32 (3 VALU per packed pair).
// ---------------------------------------------------------------------------

typedef __attribute__((ext_vector_type(16))) __bf16        bf16x16;
typedef __attribute__((ext_vector_type(8)))  float         f32x8;
typedef __attribute__((ext_vector_type(8)))  unsigned short u16x8;

union BF16Frag {
    bf16x16 b;
    u16x8   u[2];
};

#define NLAYER 16
#define NB 4
#define NS 512
#define NH 2048
#define NF 512
#define NM (NB * NS)          // 2048 rows per layer

#define TM 128                // workgroup C-tile rows
#define TN 128                // workgroup C-tile cols
#define TK 64                 // K step staged in LDS
#define LDT (TK + 8)          // padded LDS row stride (bf16 elems) -> bank spread
#define NTHREADS 256          // 8 wave32
#define CHUNKS ((TM * TK / 4) / NTHREADS)   // 8 float4-chunks per thread per tile

// f32 pair -> packed bf16x2. Round-to-nearest (ties away): add 0x8000 to each,
// then v_perm_b32 selects the two high halves: {b[31:16], a[31:16]}.
static __device__ __forceinline__ unsigned int pack2_bf16(float a, float b) {
    unsigned int ua = __float_as_uint(a) + 0x8000u;
    unsigned int ub = __float_as_uint(b) + 0x8000u;
    return __builtin_amdgcn_perm(ub, ua, 0x07060302u);
}

// Register prefetch for one (A,B) tile pair: 16 x float4 per thread.
struct Prefetch {
    float4 a[CHUNKS];
    float4 b[CHUNKS];
};

static __device__ __forceinline__ void fetch_tiles(
    const float* __restrict__ srcA, size_t ldA,
    const float* __restrict__ srcB, size_t ldB,
    int tid, Prefetch& p)
{
#pragma unroll
    for (int it = 0; it < CHUNKS; ++it) {
        int q = tid + it * NTHREADS;
        int r = q >> 4;               // TK/4 == 16 float4-chunks per row
        int c = (q & 15) << 2;
        p.a[it] = *(const float4*)(srcA + (size_t)r * ldA + c);
        p.b[it] = *(const float4*)(srcB + (size_t)r * ldB + c);
    }
}

static __device__ __forceinline__ void store_tiles(
    const Prefetch& p, int tid,
    unsigned short* __restrict__ As, unsigned short* __restrict__ Bs)
{
#pragma unroll
    for (int it = 0; it < CHUNKS; ++it) {
        int q = tid + it * NTHREADS;
        int r = q >> 4;
        int c = (q & 15) << 2;
        uint2 pa, pb;
        pa.x = pack2_bf16(p.a[it].x, p.a[it].y);
        pa.y = pack2_bf16(p.a[it].z, p.a[it].w);
        pb.x = pack2_bf16(p.b[it].x, p.b[it].y);
        pb.y = pack2_bf16(p.b[it].z, p.b[it].w);
        *(uint2*)(As + r * LDT + c) = pa;
        *(uint2*)(Bs + r * LDT + c) = pb;
    }
}

// One TK-deep MMA step for a wave: 2(M) x 4(N) tiles of v_wmma_f32_16x16x32_bf16.
// A-frag per lane: K chunks [g*8, g*8+8) and [16+g*8, 16+g*8+8) of row m (ISA 7.12.2).
// B-frag per lane: K chunks [g*8, g*8+8) and [16+g*8, 16+g*8+8) at n = lane&15.
static __device__ __forceinline__ void mma_step(
    const unsigned short* __restrict__ As, const unsigned short* __restrict__ Bs,
    int wm, int wn, int g, int ln, f32x8 acc[2][4])
{
#pragma unroll
    for (int ks = 0; ks < TK; ks += 32) {
        BF16Frag a[2];
#pragma unroll
        for (int i = 0; i < 2; ++i) {
            const unsigned short* ap = As + (wm * 32 + i * 16 + ln) * LDT + ks + g * 8;
            a[i].u[0] = *(const u16x8*)(ap);
            a[i].u[1] = *(const u16x8*)(ap + 16);
        }
        BF16Frag b[4];
#pragma unroll
        for (int j = 0; j < 4; ++j) {
            const unsigned short* bp = Bs + (wn * 64 + j * 16 + ln) * LDT + ks + g * 8;
            b[j].u[0] = *(const u16x8*)(bp);
            b[j].u[1] = *(const u16x8*)(bp + 16);
        }
#pragma unroll
        for (int i = 0; i < 2; ++i)
#pragma unroll
            for (int j = 0; j < 4; ++j)
                acc[i][j] = __builtin_amdgcn_wmma_f32_16x16x32_bf16(
                    false, a[i].b, false, b[j].b, (short)0, acc[i][j], false, false);
    }
}

// ---------------------------------------------------------------------------
// Encode: feats[l, m, f] = relu(sum_h resid[l, m, h] * enc_w[l, f, h])
// grid = (NM/TM, NF/TN, NLAYER), block = 256
// ---------------------------------------------------------------------------
__global__ __launch_bounds__(NTHREADS) void clt_encode_kernel(
    const float* __restrict__ resid, const float* __restrict__ enc_w,
    float* __restrict__ feats)
{
    __shared__ unsigned short As[2][TM * LDT];
    __shared__ unsigned short Bs[2][TN * LDT];

    const int l    = blockIdx.z;
    const int row0 = blockIdx.x * TM;
    const int col0 = blockIdx.y * TN;

    const float* AG  = resid + (size_t)l * NM * NH + (size_t)row0 * NH;
    const float* BG  = enc_w + (size_t)l * NF * NH + (size_t)col0 * NH;
    float*       out = feats + (size_t)l * NM * NF;

    const int tid  = threadIdx.x;
    const int lane = tid & 31, g = lane >> 4, ln = lane & 15;
    const int wave = tid >> 5, wm = wave & 3, wn = wave >> 2;

    f32x8 acc[2][4];
#pragma unroll
    for (int i = 0; i < 2; ++i)
#pragma unroll
        for (int j = 0; j < 4; ++j) acc[i][j] = (f32x8)(0.0f);

    const int NSTEPS = NH / TK;   // 32
    Prefetch pf;
    fetch_tiles(AG, NH, BG, NH, tid, pf);
    store_tiles(pf, tid, As[0], Bs[0]);

    for (int st = 0; st < NSTEPS; ++st) {
        const int  cur     = st & 1;
        const bool hasNext = (st + 1) < NSTEPS;
        if (hasNext)
            fetch_tiles(AG + (st + 1) * TK, NH, BG + (st + 1) * TK, NH, tid, pf);
        __syncthreads();
        mma_step(As[cur], Bs[cur], wm, wn, g, ln, acc);
        if (hasNext)
            store_tiles(pf, tid, As[cur ^ 1], Bs[cur ^ 1]);
    }

    // C/D layout (ISA 7.12.2): VGPR v, lane group g: row = g*8 + v, col = lane&15
#pragma unroll
    for (int i = 0; i < 2; ++i)
#pragma unroll
        for (int j = 0; j < 4; ++j) {
            const int col = col0 + wn * 64 + j * 16 + ln;
#pragma unroll
            for (int v = 0; v < 8; ++v) {
                const int row = row0 + wm * 32 + i * 16 + g * 8 + v;
                out[(size_t)row * NF + col] = fmaxf(acc[i][j][v], 0.0f);  // JumpReLU(0)
            }
        }
}

// ---------------------------------------------------------------------------
// Decode: recon[t, m, h] = sum_{s<=t} sum_f feats[s, m, f] * dec_w[s, t, h, f]
// grid = (NM/TM, NH/TN, NLAYER), block = 256
// The (s, kk) loops are flattened into one pipelined step index.
// ---------------------------------------------------------------------------
static __device__ __forceinline__ void dec_step_ptrs(
    const float* __restrict__ feats, const float* __restrict__ dec_w,
    int t, int row0, int col0, int st,
    const float*& pa, const float*& pb)
{
    const int s  = st >> 3;               // NF/TK == 8 steps per source layer
    const int kk = (st & 7) * TK;
    pa = feats + (size_t)s * NM * NF + (size_t)row0 * NF + kk;
    pb = dec_w + ((size_t)s * NLAYER + t) * (size_t)NH * NF + (size_t)col0 * NF + kk;
}

__global__ __launch_bounds__(NTHREADS) void clt_decode_kernel(
    const float* __restrict__ feats, const float* __restrict__ dec_w,
    float* __restrict__ recon)
{
    __shared__ unsigned short As[2][TM * LDT];
    __shared__ unsigned short Bs[2][TN * LDT];

    const int t    = blockIdx.z;
    const int row0 = blockIdx.x * TM;
    const int col0 = blockIdx.y * TN;   // over H

    const int tid  = threadIdx.x;
    const int lane = tid & 31, g = lane >> 4, ln = lane & 15;
    const int wave = tid >> 5, wm = wave & 3, wn = wave >> 2;

    f32x8 acc[2][4];
#pragma unroll
    for (int i = 0; i < 2; ++i)
#pragma unroll
        for (int j = 0; j < 4; ++j) acc[i][j] = (f32x8)(0.0f);

    const int NSTEPS = (t + 1) * (NF / TK);   // triangular: s = 0..t

    Prefetch pf;
    const float *pa, *pb;
    dec_step_ptrs(feats, dec_w, t, row0, col0, 0, pa, pb);
    fetch_tiles(pa, NF, pb, NF, tid, pf);
    store_tiles(pf, tid, As[0], Bs[0]);

    for (int st = 0; st < NSTEPS; ++st) {
        const int  cur     = st & 1;
        const bool hasNext = (st + 1) < NSTEPS;
        if (hasNext) {
            dec_step_ptrs(feats, dec_w, t, row0, col0, st + 1, pa, pb);
            fetch_tiles(pa, NF, pb, NF, tid, pf);
        }
        __syncthreads();
        mma_step(As[cur], Bs[cur], wm, wn, g, ln, acc);
        if (hasNext)
            store_tiles(pf, tid, As[cur ^ 1], Bs[cur ^ 1]);
    }

#pragma unroll
    for (int i = 0; i < 2; ++i)
#pragma unroll
        for (int j = 0; j < 4; ++j) {
            const int col = col0 + wn * 64 + j * 16 + ln;
#pragma unroll
            for (int v = 0; v < 8; ++v) {
                const int row = row0 + wm * 32 + i * 16 + g * 8 + v;
                recon[((size_t)t * NM + row) * NH + col] = acc[i][j][v];
            }
        }
}

extern "C" void kernel_launch(void* const* d_in, const int* in_sizes, int n_in,
                              void* d_out, int out_size, void* d_ws, size_t ws_size,
                              hipStream_t stream) {
    (void)in_sizes; (void)n_in; (void)out_size; (void)d_ws; (void)ws_size;
    const float* resid = (const float*)d_in[0];
    const float* enc_w = (const float*)d_in[1];
    const float* dec_w = (const float*)d_in[2];

    float* feats = (float*)d_out;                                  // [L,B,S,F]
    float* recon = feats + (size_t)NLAYER * NM * NF;               // [L,B,S,H]

    clt_encode_kernel<<<dim3(NM / TM, NF / TN, NLAYER), dim3(NTHREADS), 0, stream>>>(
        resid, enc_w, feats);
    clt_decode_kernel<<<dim3(NM / TM, NH / TN, NLAYER), dim3(NTHREADS), 0, stream>>>(
        feats, dec_w, recon);
}
